// INSID3_14577119002796
// MI455X (gfx1250) — compile-verified, matrix operand order checked by reading, not run on hardware
//
#include <hip/hip_runtime.h>

typedef __bf16 bf16_t;
typedef __attribute__((ext_vector_type(16))) __bf16 v16bf;
typedef __attribute__((ext_vector_type(8)))  __bf16 v8bf;
typedef __attribute__((ext_vector_type(8)))  float  v8f;

#define HW     4096      // 64*64
#define C_DIM  1024
#define SVD    500
#define SVDP   512       // padded
#define NPIX   12288     // 3 frames * 4096 pixels

// ---------------------------------------------------------------------------
// Fragment loader: row-major bf16 row (K contiguous). Per the CDNA5 16-bit
// A-matrix layout, lane holds K = {k0..k0+7, k0+16..k0+23} with k0 including
// the +8 shift for lanes 16..31. Two 16B loads -> one v16bf fragment.
// ---------------------------------------------------------------------------
__device__ __forceinline__ v16bf load_frag(const bf16_t* __restrict__ row, int k0) {
    v8bf lo = *(const v8bf*)(row + k0);
    v8bf hi = *(const v8bf*)(row + k0 + 16);
    return __builtin_shufflevector(lo, hi, 0,1,2,3,4,5,6,7,8,9,10,11,12,13,14,15);
}

// ---------------------------------------------------------------------------
// Generic C = A(MxK) * B(NxK)^T, bf16 operands, fp32 accumulate via WMMA.
// MODE 0: store fp32 C      (sim)
// MODE 1: store bf16 C      (coef)
// MODE 2: store fp32 (sub - C)   (Xd = Xn - recon)
// Block: 256 threads = 8 waves; block tile 64(M) x 128(N); wave tile 32x32.
// ---------------------------------------------------------------------------
template <int MODE>
__global__ __launch_bounds__(256)
void gemm_bf16_wmma(const bf16_t* __restrict__ A, const bf16_t* __restrict__ B,
                    int M, int N, int K,
                    float* __restrict__ outF, bf16_t* __restrict__ outH,
                    const bf16_t* __restrict__ sub)
{
    const int lane   = threadIdx.x & 31;
    const int wave   = threadIdx.x >> 5;
    const int laneHi = lane >> 4;           // 0 or 1
    const int lr     = lane & 15;
    const int m0     = blockIdx.y * 64 + (wave & 1) * 32;
    const int n0     = blockIdx.x * 128 + (wave >> 1) * 32;

    const bf16_t* a0 = A + (size_t)(m0 + lr) * K;
    const bf16_t* a1 = a0 + (size_t)16 * K;
    const bf16_t* b0 = B + (size_t)(n0 + lr) * K;
    const bf16_t* b1 = b0 + (size_t)16 * K;

    v8f c00 = {}; v8f c01 = {}; v8f c10 = {}; v8f c11 = {};

    for (int kk = 0; kk < K; kk += 32) {
        const int k0 = kk + laneHi * 8;
        v16bf fa0 = load_frag(a0, k0);
        v16bf fa1 = load_frag(a1, k0);
        v16bf fb0 = load_frag(b0, k0);
        v16bf fb1 = load_frag(b1, k0);
        c00 = __builtin_amdgcn_wmma_f32_16x16x32_bf16(false, fa0, false, fb0, (short)0, c00, false, false);
        c01 = __builtin_amdgcn_wmma_f32_16x16x32_bf16(false, fa0, false, fb1, (short)0, c01, false, false);
        c10 = __builtin_amdgcn_wmma_f32_16x16x32_bf16(false, fa1, false, fb0, (short)0, c10, false, false);
        c11 = __builtin_amdgcn_wmma_f32_16x16x32_bf16(false, fa1, false, fb1, (short)0, c11, false, false);
    }

    // C/D layout: VGPR r -> M = r + 8*laneHi, N = lane%16
    #pragma unroll
    for (int r = 0; r < 8; ++r) {
        const int mA = m0 + r + 8 * laneHi;
        const int nA = n0 + lr;
        const size_t i00 = (size_t)mA * N + nA;
        const size_t i01 = (size_t)mA * N + nA + 16;
        const size_t i10 = (size_t)(mA + 16) * N + nA;
        const size_t i11 = (size_t)(mA + 16) * N + nA + 16;
        if (MODE == 0) {
            outF[i00] = c00[r]; outF[i01] = c01[r];
            outF[i10] = c10[r]; outF[i11] = c11[r];
        } else if (MODE == 1) {
            outH[i00] = (bf16_t)c00[r]; outH[i01] = (bf16_t)c01[r];
            outH[i10] = (bf16_t)c10[r]; outH[i11] = (bf16_t)c11[r];
        } else {
            outF[i00] = (float)sub[i00] - c00[r];
            outF[i01] = (float)sub[i01] - c01[r];
            outF[i10] = (float)sub[i10] - c10[r];
            outF[i11] = (float)sub[i11] - c11[r];
        }
    }
}

// ---------------------------------------------------------------------------
// Pass 1a: per-pixel channel L2 norm of raw fmaps (channel-major layout).
// Thread per pixel; lanes read consecutive p -> coalesced.
// ---------------------------------------------------------------------------
__global__ void k_norms1(const float* __restrict__ f, float* __restrict__ rinv1) {
    int id = blockIdx.x * blockDim.x + threadIdx.x;     // 0..NPIX-1
    if (id >= NPIX) return;
    int t = id >> 12, p = id & 4095;
    const float* base = f + (size_t)t * C_DIM * HW + p;
    float s = 0.f;
    for (int c = 0; c < C_DIM; ++c) { float v = base[(size_t)c * HW]; s += v * v; }
    rinv1[id] = 1.0f / fmaxf(sqrtf(s), 1e-12f);
}

// ---------------------------------------------------------------------------
// Pass 1b: transpose [t][c][p] -> [t][p][c] with normalization, bf16 out.
// 32x32 LDS tile transpose.
// ---------------------------------------------------------------------------
__global__ __launch_bounds__(256)
void k_transpose_norm(const float* __restrict__ f, const float* __restrict__ rinv1,
                      bf16_t* __restrict__ Xn)
{
    __shared__ float tile[32][33];
    const int t = blockIdx.z, pT = blockIdx.y * 32, cT = blockIdx.x * 32;
    #pragma unroll
    for (int i = 0; i < 4; ++i) {
        int r = threadIdx.x + i * 256;
        int cc = r >> 5, pp = r & 31;
        tile[cc][pp] = f[((size_t)t * C_DIM + (cT + cc)) * HW + pT + pp];
    }
    __syncthreads();
    #pragma unroll
    for (int i = 0; i < 4; ++i) {
        int r = threadIdx.x + i * 256;
        int pp = r >> 5, cc = r & 31;
        float rv = rinv1[t * HW + pT + pp];
        Xn[((size_t)t * HW + pT + pp) * C_DIM + cT + cc] = (bf16_t)(tile[cc][pp] * rv);
    }
}

// ---------------------------------------------------------------------------
// basisN (1024 x 512, row=c, K=svd contiguous) and basisT (512 x 1024,
// row=svd, K=c contiguous), zero padded svd 500 -> 512, both bf16.
// ---------------------------------------------------------------------------
__global__ void k_basis_prep(const float* __restrict__ basis,
                             bf16_t* __restrict__ basisN, bf16_t* __restrict__ basisT)
{
    int idx = blockIdx.x * 256 + threadIdx.x;   // 0 .. 1024*512-1
    int c = idx >> 9, k = idx & 511;
    float v = (k < SVD) ? basis[(size_t)c * SVD + k] : 0.f;
    basisN[(size_t)c * SVDP + k] = (bf16_t)v;
    basisT[(size_t)k * C_DIM + c] = (bf16_t)v;
}

// ---------------------------------------------------------------------------
// Per-pixel renorm of Xd and bf16 feature output (wave per pixel, wave32).
// ---------------------------------------------------------------------------
__global__ __launch_bounds__(256)
void k_finalize(const float* __restrict__ Xd, float* __restrict__ rinv2,
                bf16_t* __restrict__ Fd)
{
    const int wave = threadIdx.x >> 5, lane = threadIdx.x & 31;
    const int pix = blockIdx.x * 8 + wave;
    const float* row = Xd + (size_t)pix * C_DIM;
    float s = 0.f;
    for (int c = lane; c < C_DIM; c += 32) { float v = row[c]; s += v * v; }
    #pragma unroll
    for (int o = 16; o > 0; o >>= 1) s += __shfl_xor(s, o, 32);
    const float rv = 1.0f / fmaxf(sqrtf(s), 1e-12f);
    if (lane == 0) rinv2[pix] = rv;
    bf16_t* orow = Fd + (size_t)pix * C_DIM;
    for (int c = lane; c < C_DIM; c += 32) orow[c] = (bf16_t)(row[c] * rv);
}

__global__ void k_counts(const int* __restrict__ mask, float* __restrict__ counts) {
    __shared__ float sm[256];
    const int s = blockIdx.x;
    float acc = 0.f;
    for (int p = threadIdx.x; p < HW; p += 256) acc += (mask[s * HW + p] != 0) ? 1.f : 0.f;
    sm[threadIdx.x] = acc; __syncthreads();
    for (int o = 128; o > 0; o >>= 1) {
        if (threadIdx.x < o) sm[threadIdx.x] += sm[threadIdx.x + o];
        __syncthreads();
    }
    if (threadIdx.x == 0) counts[s] = fmaxf(sm[0], 1.0f);
}

__global__ void k_protos(const float* __restrict__ Xd, const float* __restrict__ rinv2,
                         const int* __restrict__ mask, const float* __restrict__ counts,
                         float* __restrict__ protos)
{
    int idx = blockIdx.x * 256 + threadIdx.x;   // 0..2047
    int s = idx >> 10, c = idx & 1023;
    float acc = 0.f;
    for (int p = 0; p < HW; ++p) {
        if (mask[s * HW + p] != 0)
            acc += Xd[((size_t)(s * HW + p)) * C_DIM + c] * rinv2[s * HW + p];
    }
    protos[idx] = acc / counts[s];
}

__global__ __launch_bounds__(1024)
void k_prototype(const float* __restrict__ protos, float* __restrict__ proto) {
    __shared__ float sm[1024];
    const int c = threadIdx.x;
    float m = 0.5f * (protos[c] + protos[1024 + c]);
    sm[c] = m * m; __syncthreads();
    for (int o = 512; o > 0; o >>= 1) {
        if (c < o) sm[c] += sm[c + o];
        __syncthreads();
    }
    proto[c] = m / fmaxf(sqrtf(sm[0]), 1e-12f);
}

__global__ void k_simfwd(const float* __restrict__ Xd, const float* __restrict__ rinv2,
                         const float* __restrict__ proto, float* __restrict__ out)
{
    const int wave = threadIdx.x >> 5, lane = threadIdx.x & 31;
    const int p = blockIdx.x * 8 + wave;
    const float* row = Xd + ((size_t)(2 * HW + p)) * C_DIM;
    float s = 0.f;
    for (int c = lane; c < C_DIM; c += 32) s += row[c] * proto[c];
    #pragma unroll
    for (int o = 16; o > 0; o >>= 1) s += __shfl_xor(s, o, 32);
    if (lane == 0) out[p] = s * rinv2[2 * HW + p];
}

__global__ void k_votes(const float* __restrict__ sim, const int* __restrict__ mask,
                        float* __restrict__ out)
{
    const int pt = blockIdx.x * 256 + threadIdx.x;  // target pixel
    int v = 0;
    for (int s = 0; s < 2; ++s) {
        float best = -3.4e38f; int bi = 0;
        const float* col = sim + (size_t)s * HW * HW + pt;
        for (int pr = 0; pr < HW; ++pr) {
            float val = col[(size_t)pr * HW];
            if (val > best) { best = val; bi = pr; }   // first-occurrence argmax
        }
        v += (mask[s * HW + bi] != 0) ? 1 : 0;
    }
    out[pt] = (float)v;
}

// ---------------------------------------------------------------------------
extern "C" void kernel_launch(void* const* d_in, const int* in_sizes, int n_in,
                              void* d_out, int out_size, void* d_ws, size_t ws_size,
                              hipStream_t stream)
{
    (void)in_sizes; (void)n_in; (void)out_size; (void)ws_size;
    const float* fmaps = (const float*)d_in[0];
    const int*   masks = (const int*)d_in[1];
    const float* basis = (const float*)d_in[2];
    float* out = (float*)d_out;

    char* ws = (char*)d_ws;
    size_t off = 0;
    auto walloc = [&](size_t bytes) -> void* {
        void* p = ws + off; off += (bytes + 255) & ~(size_t)255; return p;
    };
    bf16_t* Xn     = (bf16_t*)walloc((size_t)NPIX * C_DIM * 2);  // reused as Fd later
    float*  rinv1  = (float*) walloc((size_t)NPIX * 4);
    bf16_t* basisT = (bf16_t*)walloc((size_t)SVDP * C_DIM * 2);
    bf16_t* basisN = (bf16_t*)walloc((size_t)C_DIM * SVDP * 2);
    bf16_t* coef   = (bf16_t*)walloc((size_t)NPIX * SVDP * 2);
    float*  Xd     = (float*) walloc((size_t)NPIX * C_DIM * 4);
    float*  rinv2  = (float*) walloc((size_t)NPIX * 4);
    float*  protos = (float*) walloc((size_t)2 * C_DIM * 4);
    float*  counts = (float*) walloc(2 * 4);
    float*  proto  = (float*) walloc((size_t)C_DIM * 4);

    // 1) L2 normalize along channels + transpose to pixel-major bf16
    k_norms1<<<48, 256, 0, stream>>>(fmaps, rinv1);
    k_transpose_norm<<<dim3(C_DIM / 32, HW / 32, 3), 256, 0, stream>>>(fmaps, rinv1, Xn);
    k_basis_prep<<<(C_DIM * SVDP) / 256, 256, 0, stream>>>(basis, basisN, basisT);

    // 2) coef = Xn * basis  (12288 x 512, K=1024) -> bf16
    gemm_bf16_wmma<1><<<dim3(SVDP / 128, NPIX / 64), 256, 0, stream>>>(
        Xn, basisT, NPIX, SVDP, C_DIM, nullptr, coef, nullptr);

    // 3) Xd = Xn - coef * basis^T  (12288 x 1024, K=512) -> fp32
    gemm_bf16_wmma<2><<<dim3(C_DIM / 128, NPIX / 64), 256, 0, stream>>>(
        coef, basisN, NPIX, C_DIM, SVDP, Xd, nullptr, Xn);

    // 4) renormalize -> Fd (bf16, reuses Xn region), rinv2
    bf16_t* Fd = Xn;
    k_finalize<<<NPIX / 8, 256, 0, stream>>>(Xd, rinv2, Fd);

    // 5) sim = Fd_refs * Fd_tgt^T  (8192 x 4096, K=1024) -> d_out[0..]
    gemm_bf16_wmma<0><<<dim3(HW / 128, (2 * HW) / 64), 256, 0, stream>>>(
        Fd, Fd + (size_t)2 * HW * C_DIM, 2 * HW, HW, C_DIM, out, nullptr, nullptr);

    // 6) prototypes, sim_fwd, votes
    k_counts<<<2, 256, 0, stream>>>(masks, counts);
    k_protos<<<8, 256, 0, stream>>>(Xd, rinv2, masks, counts, protos);
    k_prototype<<<1, 1024, 0, stream>>>(protos, proto);
    k_simfwd<<<HW / 8, 256, 0, stream>>>(Xd, rinv2, proto, out + (size_t)2 * HW * HW);
    k_votes<<<HW / 256, 256, 0, stream>>>(out, masks, out + (size_t)2 * HW * HW + HW);
}
